// RGCN_54631984005709
// MI455X (gfx1250) — matrix operand order repaired and checked by reading, not compiled, not run on hardware
//
#include <hip/hip_runtime.h>
#include <hip/hip_bf16.h>
#include <math.h>

// ---------------- problem constants ----------------
static constexpr int    NN = 100000;     // nodes
static constexpr int    NE = 800000;     // edges
static constexpr int    NG = 2048;       // graphs
static constexpr int    D  = 64;         // hidden dim
static constexpr int    R  = 5;          // relations
static constexpr int    PROP = 6;
static constexpr int    S2S  = 6;
static constexpr int    MT = NN / 16;    // 6250 row tiles (exact)

// ---------------- WMMA vector types ----------------
typedef __attribute__((ext_vector_type(16))) __bf16 bf16x16;
typedef __attribute__((ext_vector_type(8)))  float  f32x8;

// ---------------- workspace layout (bytes) ----------------
static constexpr size_t OFF_H0    = 0;
static constexpr size_t OFF_H1    = OFF_H0   + (size_t)NN * D * 4;   // f32 h ping
static constexpr size_t OFF_HBF   = OFF_H1   + (size_t)NN * D * 4;   // f32 h pong
static constexpr size_t OFF_HREL  = OFF_HBF  + (size_t)NN * D * 2;   // bf16 h
static constexpr size_t OFF_AGG   = OFF_HREL + (size_t)R * NN * D * 2; // bf16 h_rel
static constexpr size_t OFF_CNT   = OFF_AGG  + (size_t)NN * D * 4;   // f32 agg
static constexpr size_t OFF_WFRAG = OFF_CNT  + (size_t)NN * 4;       // f32 cnt
static constexpr size_t OFF_ROOTF = OFF_WFRAG + (size_t)R * 8 * 512 * 2; // bf16 W frags
static constexpr size_t OFF_QSTAR = OFF_ROOTF + (size_t)8 * 512 * 2;     // bf16 root frags
static constexpr size_t OFF_HX    = OFF_QSTAR + (size_t)NG * 2 * D * 4;
static constexpr size_t OFF_CX    = OFF_HX    + (size_t)NG * D * 4;
static constexpr size_t OFF_GATES = OFF_CX    + (size_t)NG * D * 4;
static constexpr size_t OFF_EBUF  = OFF_GATES + (size_t)NG * 4 * D * 4;
static constexpr size_t OFF_M     = OFF_EBUF  + (size_t)NN * 4;
static constexpr size_t OFF_S     = OFF_M     + (size_t)NG * 4;
static constexpr size_t OFF_RBUF  = OFF_S     + (size_t)NG * 4;
static constexpr size_t OFF_L1O   = OFF_RBUF  + (size_t)NG * D * 4;

// ---------------- helpers ----------------
__global__ void fill_kernel(float* __restrict__ p, float v, int n) {
  int i = blockIdx.x * blockDim.x + threadIdx.x;
  if (i < n) p[i] = v;
}

__device__ __forceinline__ void atomicMaxF(float* addr, float v) {
  // IEEE ordering trick: positive floats order as ints, negative as reversed uints.
  if (v >= 0.0f) atomicMax((int*)addr, __float_as_int(v));
  else           atomicMin((unsigned int*)addr, __float_as_uint(v));
}

__device__ __forceinline__ float sigf(float x) { return 1.0f / (1.0f + __expf(-x)); }

// ---------------- weight prep: W[r]=att@basis, shuffle into B-fragment layout ----------------
// B fragment layout for V_WMMA_F32_16X16X32_BF16 (32x16 bf16 B):
//   lane = (khalf<<4) | Ncol ; element j -> K = ktile*32 + (j>>3)*16 + khalf*8 + (j&7)
// fragment f=(r*4+nt)*2+kt stored lane-major: frag[f*512 + lane*16 + j]
__global__ void prep_weights_kernel(const float* __restrict__ att,
                                    const float* __restrict__ basis,
                                    const float* __restrict__ root,
                                    __bf16* __restrict__ wfrag,
                                    __bf16* __restrict__ rootfrag) {
  __shared__ float Wc[R * D * D];   // 80 KB LDS (<=320KB per WGP on CDNA5)
  for (int i = threadIdx.x; i < R * D * D; i += blockDim.x) {
    int r = i / (D * D), de = i % (D * D);
    float acc = 0.0f;
    #pragma unroll
    for (int b = 0; b < R; ++b) acc += att[r * R + b] * basis[b * D * D + de];
    Wc[i] = acc;
  }
  __syncthreads();
  for (int i = threadIdx.x; i < R * 8 * 512; i += blockDim.x) {
    int f = i / 512, lj = i % 512, lane = lj / 16, j = lj % 16;
    int kt = f & 1, nt = (f >> 1) & 3, r = f >> 3;
    int K = kt * 32 + (j >> 3) * 16 + (lane >> 4) * 8 + (j & 7);
    int Ncol = nt * 16 + (lane & 15);
    wfrag[i] = (__bf16)Wc[r * D * D + K * D + Ncol];
  }
  for (int i = threadIdx.x; i < 8 * 512; i += blockDim.x) {
    int f = i / 512, lj = i % 512, lane = lj / 16, j = lj % 16;
    int kt = f & 1, nt = f >> 1;
    int K = kt * 32 + (j >> 3) * 16 + (lane >> 4) * 8 + (j & 7);
    int Ncol = nt * 16 + (lane & 15);
    rootfrag[i] = (__bf16)root[K * D + Ncol];
  }
}

// ---------------- input projection: h = relu(x @ lin0_w + b) ----------------
__global__ void lin0_kernel(const float* __restrict__ x, const float* __restrict__ w,
                            const float* __restrict__ b, float* __restrict__ h) {
  int tid = blockIdx.x * blockDim.x + threadIdx.x;
  if (tid >= NN * D) return;
  int n = tid >> 6, d = tid & 63;
  float acc = b[d];
  #pragma unroll
  for (int k = 0; k < 15; ++k) acc += x[n * 15 + k] * w[k * D + d];
  h[tid] = fmaxf(acc, 0.0f);
}

__global__ void cast_bf16_kernel(const float* __restrict__ h, __bf16* __restrict__ hbf) {
  int i = blockIdx.x * blockDim.x + threadIdx.x;
  if (i < NN * D) hbf[i] = (__bf16)h[i];
}

// ---------------- load A fragment (ISA 16-bit A 16x32 layout) ----------------
__device__ __forceinline__ void load_a_frags(const __bf16* __restrict__ hbf,
                                             int mrow, int lane,
                                             bf16x16& a0, bf16x16& a1) {
  const __bf16* p = hbf + (size_t)mrow * D;
  const int grp = (lane >> 4) * 8;
  #pragma unroll
  for (int j = 0; j < 8; ++j) {
    a0[j]     = p[grp + j];            // K =  grp..grp+7
    a0[8 + j] = p[16 + grp + j];       // K = 16+grp..
    a1[j]     = p[32 + grp + j];       // K = 32+grp..
    a1[8 + j] = p[48 + grp + j];       // K = 48+grp..
  }
}

// ---------------- h_rel[r] = h @ W[r]  (one wave per (relation, 16-row tile)) ----------------
__global__ void hrel_wmma_kernel(const __bf16* __restrict__ hbf,
                                 const __bf16* __restrict__ wfrag,
                                 __bf16* __restrict__ hrel) {
  int wave = blockIdx.x * (blockDim.x >> 5) + (threadIdx.x >> 5);
  int lane = threadIdx.x & 31;
  if (wave >= MT * R) return;                 // wave-uniform exit keeps EXEC full for WMMA
  int r = wave / MT, mtile = wave % MT;
  int mrow = mtile * 16 + (lane & 15);
  int halfsel = (lane >> 4) * 8;

  bf16x16 a0, a1;
  load_a_frags(hbf, mrow, lane, a0, a1);

  #pragma unroll
  for (int nt = 0; nt < 4; ++nt) {
    const __bf16* bbase = wfrag + (size_t)((r * 4 + nt) * 2) * 512;
    bf16x16 b0 = *(const bf16x16*)(bbase + lane * 16);
    bf16x16 b1 = *(const bf16x16*)(bbase + 512 + lane * 16);
    f32x8 c = {};
    c = __builtin_amdgcn_wmma_f32_16x16x32_bf16(false, a0, false, b0, (short)0, c, false, false);
    c = __builtin_amdgcn_wmma_f32_16x16x32_bf16(false, a1, false, b1, (short)0, c, false, false);
    __bf16* out = hrel + ((size_t)r * NN + (size_t)mtile * 16) * D + nt * 16 + (lane & 15);
    #pragma unroll
    for (int v = 0; v < 8; ++v)
      out[(size_t)(v + halfsel) * D] = (__bf16)c[v];
  }
}

// ---------------- edge scatter: agg[dst] += h_rel[et][src]; cnt[dst] += 1 ----------------
__global__ void edge_scatter_kernel(const int* __restrict__ edge_index,
                                    const int* __restrict__ edge_type,
                                    const __bf16* __restrict__ hrel,
                                    float* __restrict__ agg,
                                    float* __restrict__ cnt) {
  int tid = blockIdx.x * blockDim.x + threadIdx.x;     // NE * 8 threads
  if (tid >= NE * 8) return;
  int e = tid >> 3, c0 = (tid & 7) * 8;
  int src = edge_index[e];
  int dst = edge_index[NE + e];
  int et  = edge_type[e];
  const __bf16* msg = hrel + ((size_t)et * NN + (size_t)src) * D + c0;
  float* ag = agg + (size_t)dst * D + c0;
  #pragma unroll
  for (int j = 0; j < 8; ++j) atomicAdd(ag + j, (float)msg[j]);
  if (c0 == 0) atomicAdd(cnt + dst, 1.0f);
}

// ---------------- node update: h' = relu(agg/max(cnt,1) + h@root + b) ----------------
__global__ void node_wmma_kernel(const __bf16* __restrict__ hbf,
                                 const __bf16* __restrict__ rootfrag,
                                 const float* __restrict__ agg,
                                 const float* __restrict__ cnt,
                                 const float* __restrict__ conv_b,
                                 float* __restrict__ hout) {
  int wave = blockIdx.x * (blockDim.x >> 5) + (threadIdx.x >> 5);
  int lane = threadIdx.x & 31;
  if (wave >= MT) return;
  int mtile = wave;
  int mrow = mtile * 16 + (lane & 15);
  int halfsel = (lane >> 4) * 8;

  bf16x16 a0, a1;
  load_a_frags(hbf, mrow, lane, a0, a1);

  #pragma unroll
  for (int nt = 0; nt < 4; ++nt) {
    const __bf16* bbase = rootfrag + (size_t)(nt * 2) * 512;
    bf16x16 b0 = *(const bf16x16*)(bbase + lane * 16);
    bf16x16 b1 = *(const bf16x16*)(bbase + 512 + lane * 16);
    f32x8 c = {};
    c = __builtin_amdgcn_wmma_f32_16x16x32_bf16(false, a0, false, b0, (short)0, c, false, false);
    c = __builtin_amdgcn_wmma_f32_16x16x32_bf16(false, a1, false, b1, (short)0, c, false, false);
    int col = nt * 16 + (lane & 15);
    #pragma unroll
    for (int v = 0; v < 8; ++v) {
      int row = mtile * 16 + v + halfsel;
      size_t idx = (size_t)row * D + col;
      float denom = fmaxf(cnt[row], 1.0f);
      float val = c[v] + agg[idx] / denom + conv_b[col];
      hout[idx] = fmaxf(val, 0.0f);
    }
  }
}

// ---------------- Set2Set ----------------
__global__ void lstm_gates_kernel(const float* __restrict__ qstar, const float* __restrict__ hx,
                                  const float* __restrict__ w_ih, const float* __restrict__ w_hh,
                                  const float* __restrict__ b_ih, const float* __restrict__ b_hh,
                                  float* __restrict__ gates) {
  int g = blockIdx.x;          // 2048
  int j = threadIdx.x;         // 256
  float acc = b_ih[j] + b_hh[j];
  const float* q = qstar + (size_t)g * 2 * D;
  const float* h = hx + (size_t)g * D;
  #pragma unroll 4
  for (int k = 0; k < 2 * D; ++k) acc += q[k] * w_ih[j * 2 * D + k];
  #pragma unroll 4
  for (int k = 0; k < D; ++k)     acc += h[k] * w_hh[j * D + k];
  gates[(size_t)g * 4 * D + j] = acc;
}

__global__ void lstm_cell_kernel(const float* __restrict__ gates,
                                 float* __restrict__ hx, float* __restrict__ cx,
                                 float* __restrict__ rbuf, float* __restrict__ m,
                                 float* __restrict__ s) {
  int tid = blockIdx.x * blockDim.x + threadIdx.x;
  if (tid >= NG * D) return;
  int g = tid >> 6, d = tid & 63;
  const float* gp = gates + (size_t)g * 4 * D;
  float i  = sigf(gp[d]);
  float f  = sigf(gp[D + d]);
  float gg = tanhf(gp[2 * D + d]);
  float o  = sigf(gp[3 * D + d]);
  float c  = f * cx[tid] + i * gg;
  cx[tid]  = c;
  hx[tid]  = o * tanhf(c);
  rbuf[tid] = 0.0f;
  if (d == 0) { m[g] = -INFINITY; s[g] = 0.0f; }
}

__global__ void attn_e_kernel(const float* __restrict__ h, const float* __restrict__ hx,
                              const int* __restrict__ batch,
                              float* __restrict__ ebuf, float* __restrict__ m) {
  int n = blockIdx.x * blockDim.x + threadIdx.x;
  if (n >= NN) return;
  int g = batch[n];
  const float* hp = h + (size_t)n * D;
  const float* qp = hx + (size_t)g * D;
  float acc = 0.0f;
  #pragma unroll 8
  for (int k = 0; k < D; ++k) acc += hp[k] * qp[k];
  ebuf[n] = acc;
  atomicMaxF(m + g, acc);
}

__global__ void attn_s_kernel(const int* __restrict__ batch, float* __restrict__ ebuf,
                              const float* __restrict__ m, float* __restrict__ s) {
  int n = blockIdx.x * blockDim.x + threadIdx.x;
  if (n >= NN) return;
  int g = batch[n];
  float a = __expf(ebuf[n] - m[g]);
  ebuf[n] = a;
  atomicAdd(s + g, a);
}

__global__ void attn_r_kernel(const int* __restrict__ batch, const float* __restrict__ ebuf,
                              const float* __restrict__ s, const float* __restrict__ h,
                              float* __restrict__ rbuf) {
  int tid = blockIdx.x * blockDim.x + threadIdx.x;
  if (tid >= NN * D) return;
  int n = tid >> 6, d = tid & 63;
  int g = batch[n];
  float a = ebuf[n] / s[g];
  atomicAdd(rbuf + (size_t)g * D + d, a * h[tid]);
}

__global__ void qstar_update_kernel(const float* __restrict__ hx, const float* __restrict__ rbuf,
                                    float* __restrict__ qstar) {
  int tid = blockIdx.x * blockDim.x + threadIdx.x;
  if (tid >= NG * D) return;
  int g = tid >> 6, d = tid & 63;
  qstar[(size_t)g * 2 * D + d]     = hx[tid];
  qstar[(size_t)g * 2 * D + D + d] = rbuf[tid];
}

// ---------------- output head ----------------
__global__ void lin1_kernel(const float* __restrict__ qstar, const float* __restrict__ w,
                            const float* __restrict__ b, float* __restrict__ out) {
  int tid = blockIdx.x * blockDim.x + threadIdx.x;
  if (tid >= NG * D) return;
  int g = tid >> 6, d = tid & 63;
  float acc = b[d];
  #pragma unroll 4
  for (int k = 0; k < 2 * D; ++k) acc += qstar[(size_t)g * 2 * D + k] * w[k * D + d];
  out[tid] = fmaxf(acc, 0.0f);
}

__global__ void lin2_kernel(const float* __restrict__ l1o, const float* __restrict__ w,
                            const float* __restrict__ b, float* __restrict__ out) {
  int tid = blockIdx.x * blockDim.x + threadIdx.x;
  if (tid >= NG * 12) return;
  int g = tid / 12, j = tid % 12;
  float acc = b[j];
  #pragma unroll 8
  for (int k = 0; k < D; ++k) acc += l1o[(size_t)g * D + k] * w[k * 12 + j];
  out[tid] = acc;
}

// ---------------- host orchestration ----------------
extern "C" void kernel_launch(void* const* d_in, const int* in_sizes, int n_in,
                              void* d_out, int out_size, void* d_ws, size_t ws_size,
                              hipStream_t stream) {
  const float* x       = (const float*)d_in[0];
  const int*   eidx    = (const int*)d_in[1];
  const int*   etype   = (const int*)d_in[2];
  const int*   batch   = (const int*)d_in[3];
  const float* lin0_w  = (const float*)d_in[4];
  const float* lin0_b  = (const float*)d_in[5];
  const float* basis   = (const float*)d_in[6];
  const float* att     = (const float*)d_in[7];
  const float* root    = (const float*)d_in[8];
  const float* conv_b  = (const float*)d_in[9];
  const float* w_ih    = (const float*)d_in[10];
  const float* w_hh    = (const float*)d_in[11];
  const float* b_ih    = (const float*)d_in[12];
  const float* b_hh    = (const float*)d_in[13];
  const float* lin1_w  = (const float*)d_in[14];
  const float* lin1_b  = (const float*)d_in[15];
  const float* lin2_w  = (const float*)d_in[16];
  const float* lin2_b  = (const float*)d_in[17];

  char* ws = (char*)d_ws;
  float*  h0    = (float*)(ws + OFF_H0);
  float*  h1    = (float*)(ws + OFF_H1);
  __bf16* hbf   = (__bf16*)(ws + OFF_HBF);
  __bf16* hrel  = (__bf16*)(ws + OFF_HREL);
  float*  agg   = (float*)(ws + OFF_AGG);
  float*  cnt   = (float*)(ws + OFF_CNT);
  __bf16* wfrag = (__bf16*)(ws + OFF_WFRAG);
  __bf16* rootf = (__bf16*)(ws + OFF_ROOTF);
  float*  qstar = (float*)(ws + OFF_QSTAR);
  float*  hx    = (float*)(ws + OFF_HX);
  float*  cx    = (float*)(ws + OFF_CX);
  float*  gates = (float*)(ws + OFF_GATES);
  float*  ebuf  = (float*)(ws + OFF_EBUF);
  float*  mbuf  = (float*)(ws + OFF_M);
  float*  sbuf  = (float*)(ws + OFF_S);
  float*  rbuf  = (float*)(ws + OFF_RBUF);
  float*  l1o   = (float*)(ws + OFF_L1O);
  float*  out   = (float*)d_out;

  const int B256 = 256;

  // weights -> fragment-ready bf16
  prep_weights_kernel<<<1, B256, 0, stream>>>(att, basis, root, wfrag, rootf);

  // h = relu(x @ lin0 + b)
  lin0_kernel<<<(NN * D + B256 - 1) / B256, B256, 0, stream>>>(x, lin0_w, lin0_b, h0);

  float* h  = h0;
  float* hn = h1;
  const int hrelWaves  = MT * R;
  const int hrelBlocks = (hrelWaves + 7) / 8;
  const int nodeBlocks = (MT + 7) / 8;

  for (int step = 0; step < PROP; ++step) {
    cast_bf16_kernel<<<(NN * D + B256 - 1) / B256, B256, 0, stream>>>(h, hbf);
    // zero agg (N*D) and cnt (N) — contiguous in ws
    fill_kernel<<<(NN * D + NN + B256 - 1) / B256, B256, 0, stream>>>(agg, 0.0f, NN * D + NN);
    hrel_wmma_kernel<<<hrelBlocks, B256, 0, stream>>>(hbf, wfrag, hrel);
    edge_scatter_kernel<<<(NE * 8 + B256 - 1) / B256, B256, 0, stream>>>(eidx, etype, hrel, agg, cnt);
    node_wmma_kernel<<<nodeBlocks, B256, 0, stream>>>(hbf, rootf, agg, cnt, conv_b, hn);
    float* t = h; h = hn; hn = t;
  }

  // Set2Set: zero q_star | hx | cx (contiguous: NG*(2D + D + D) floats)
  fill_kernel<<<(NG * 4 * D + B256 - 1) / B256, B256, 0, stream>>>(qstar, 0.0f, NG * 4 * D);
  for (int it = 0; it < S2S; ++it) {
    lstm_gates_kernel<<<NG, 4 * D, 0, stream>>>(qstar, hx, w_ih, w_hh, b_ih, b_hh, gates);
    lstm_cell_kernel<<<(NG * D + B256 - 1) / B256, B256, 0, stream>>>(gates, hx, cx, rbuf, mbuf, sbuf);
    attn_e_kernel<<<(NN + B256 - 1) / B256, B256, 0, stream>>>(h, hx, batch, ebuf, mbuf);
    attn_s_kernel<<<(NN + B256 - 1) / B256, B256, 0, stream>>>(batch, ebuf, mbuf, sbuf);
    attn_r_kernel<<<(NN * D + B256 - 1) / B256, B256, 0, stream>>>(batch, ebuf, sbuf, h, rbuf);
    qstar_update_kernel<<<(NG * D + B256 - 1) / B256, B256, 0, stream>>>(hx, rbuf, qstar);
  }

  lin1_kernel<<<(NG * D + B256 - 1) / B256, B256, 0, stream>>>(qstar, lin1_w, lin1_b, l1o);
  lin2_kernel<<<(NG * 12 + B256 - 1) / B256, B256, 0, stream>>>(l1o, lin2_w, lin2_b, out);
}
